// MultiHeadAttention_37718402793745
// MI455X (gfx1250) — matrix operand order repaired and checked by reading, not compiled, not run on hardware
//
#include <hip/hip_runtime.h>
#include <hip/hip_bf16.h>
#include <math.h>

#define BATCH 2
#define SEQ   2048
#define EDIM  1024
#define NH    16
#define DH    64
#define QSCALE 0.125f   // 1/sqrt(DH)

typedef __bf16 bf16_t;
typedef __attribute__((ext_vector_type(16))) __bf16 bfx16;
typedef __attribute__((ext_vector_type(8)))  __bf16 bfx8;
typedef __attribute__((ext_vector_type(8)))  float  fx8;

// gfx1250 async global->LDS path (ASYNCcnt). Guarded so a missing builtin
// falls back to synchronous staging without breaking the compile.
#if defined(__gfx1250__) && __has_builtin(__builtin_amdgcn_global_load_async_to_lds_b128) && __has_builtin(__builtin_amdgcn_s_wait_asynccnt)
#define HAVE_ASYNC 1
#else
#define HAVE_ASYNC 0
#endif

__device__ __forceinline__ fx8 wmma_bf16(bfx16 a, bfx16 b, fx8 c) {
  // D = A(16x32 bf16) * B(32x16 bf16) + C(16x16 f32)
  return __builtin_amdgcn_wmma_f32_16x16x32_bf16(false, a, false, b, (short)0, c,
                                                 false, false);
}

// A-operand (16x32 bf16): per-lane K runs at [p, p+7] and [p+16, p+23].
__device__ __forceinline__ bfx16 frag_a(const bf16_t* p) {
  bfx8 lo = *(const bfx8*)p;
  bfx8 hi = *(const bfx8*)(p + 16);
  return __builtin_shufflevector(lo, hi, 0,1,2,3,4,5,6,7,8,9,10,11,12,13,14,15);
}
// A-operand built from an f32 tile (convert at fragment-build time).
__device__ __forceinline__ bfx16 frag_a_f32(const float* p) {
  bfx16 r;
#pragma unroll
  for (int i = 0; i < 8; i++) r[i] = (bf16_t)p[i];
#pragma unroll
  for (int i = 0; i < 8; i++) r[8 + i] = (bf16_t)p[16 + i];
  return r;
}
// B-operand (32x16 bf16): per-lane K runs at [p, p+7] and [p+16, p+23].
__device__ __forceinline__ bfx16 frag_b(const bf16_t* p) {
  bfx8 lo = *(const bfx8*)p;
  bfx8 hi = *(const bfx8*)(p + 16);
  return __builtin_shufflevector(lo, hi, 0,1,2,3,4,5,6,7,8,9,10,11,12,13,14,15);
}

#if HAVE_ASYNC
// Builtin signature (from the round-2 diagnostic): parameters are typed
// vector pointers: (int4 addrspace(1)*, int4 addrspace(3)*, imm, imm).
typedef int i32x4_vs __attribute__((vector_size(16)));
typedef __attribute__((address_space(1))) i32x4_vs g_i32x4;
typedef __attribute__((address_space(3))) i32x4_vs l_i32x4;

__device__ __forceinline__ void async_cp16(const void* g, void* l) {
  __builtin_amdgcn_global_load_async_to_lds_b128((g_i32x4*)g, (l_i32x4*)l, 0, 0);
}
// Stage one K-step tile pair: X (f32, 128x32) and WT (bf16, 128x32).
// 6 async b128 instructions per thread/wave per stage.
__device__ __forceinline__ void stage_async(const float* X, const bf16_t* WT,
                                            int m0, int n0, int kb, int tid,
                                            float (*As)[36], bf16_t (*Bs)[40]) {
#pragma unroll
  for (int i = 0; i < 4; i++) {          // 16 KB of X tile, 16 B per op
    int lin = tid + i * 256;
    int row = lin >> 3, seg = lin & 7;
    async_cp16(X + (size_t)(m0 + row) * EDIM + kb + seg * 4, &As[row][seg * 4]);
  }
#pragma unroll
  for (int i = 0; i < 2; i++) {          // 8 KB of WT tile
    int lin = tid + i * 256;
    int row = lin >> 2, seg = lin & 3;
    async_cp16(WT + (size_t)(n0 + row) * EDIM + kb + seg * 8, &Bs[row][seg * 8]);
  }
}
#else
__device__ __forceinline__ void stage_sync(const float* X, const bf16_t* WT,
                                           int m0, int n0, int kb, int tid,
                                           float (*As)[36], bf16_t (*Bs)[40]) {
#pragma unroll
  for (int i = 0; i < 4; i++) {
    int lin = tid + i * 256;
    int row = lin >> 3, seg = lin & 7;
    *(float4*)&As[row][seg * 4] =
        *(const float4*)&X[(size_t)(m0 + row) * EDIM + kb + seg * 4];
  }
#pragma unroll
  for (int i = 0; i < 2; i++) {
    int lin = tid + i * 256;
    int row = lin >> 2, seg = lin & 3;
    *(bfx8*)&Bs[row][seg * 8] =
        *(const bfx8*)&WT[(size_t)(n0 + row) * EDIM + kb + seg * 8];
  }
}
#endif

// ---------------------------------------------------------------------------
// Kernel 0: transpose + convert weights: W f32 [k][n] -> WT bf16 [n][k].
// grid = (E/64, E/64, 3), block = 256.
// ---------------------------------------------------------------------------
__global__ __launch_bounds__(256) void convert_w_kernel(
    const float* __restrict__ wq, const float* __restrict__ wk, const float* __restrict__ wv,
    bf16_t* __restrict__ wtq, bf16_t* __restrict__ wtk, bf16_t* __restrict__ wtv)
{
  __shared__ float T[64][65];
  const int which = blockIdx.z;
  const float* W = (which == 0) ? wq : (which == 1) ? wk : wv;
  bf16_t*     WT = (which == 0) ? wtq : (which == 1) ? wtk : wtv;
  const int n0 = blockIdx.x * 64, k0 = blockIdx.y * 64;
  const int tid = threadIdx.x;

#pragma unroll
  for (int i = 0; i < 4; i++) {
    int lin = tid + i * 256;           // float4 units, 16 per 64-wide row
    int r = lin >> 4, c = (lin & 15) * 4;
    float4 f = *(const float4*)&W[(size_t)(k0 + r) * EDIM + n0 + c];
    T[r][c] = f.x; T[r][c+1] = f.y; T[r][c+2] = f.z; T[r][c+3] = f.w;
  }
  __syncthreads();
#pragma unroll
  for (int i = 0; i < 4; i++) {
    int lin = tid + i * 256;
    int r = lin >> 4, c = (lin & 15) * 4;   // r = n row, c = k col
    bf16_t* dst = &WT[(size_t)(n0 + r) * EDIM + k0 + c];
    dst[0] = (bf16_t)T[c+0][r]; dst[1] = (bf16_t)T[c+1][r];
    dst[2] = (bf16_t)T[c+2][r]; dst[3] = (bf16_t)T[c+3][r];
  }
}

// ---------------------------------------------------------------------------
// Kernel 1: fused QKV projections with double-buffered async LDS staging.
// grid = (E/128, (B*S)/128, 3), block = 256.
// which = 0:Q (pre-scaled), 1:K, 2:V (stored transposed per head).
// ---------------------------------------------------------------------------
__global__ __launch_bounds__(256) void qkv_proj_kernel(
    const float* __restrict__ xq, const float* __restrict__ xk, const float* __restrict__ xv,
    const bf16_t* __restrict__ wtq, const bf16_t* __restrict__ wtk, const bf16_t* __restrict__ wtv,
    const float* __restrict__ bq, const float* __restrict__ bk, const float* __restrict__ bv,
    bf16_t* __restrict__ oq, bf16_t* __restrict__ ok, bf16_t* __restrict__ ovt)
{
  __shared__ alignas(16) float  As32[2][128][36];  // X tile, raw f32
  __shared__ alignas(16) bf16_t BsT[2][128][40];   // WT tile: [n][k] bf16

  const int which = blockIdx.z;
  const float*  X  = (which == 0) ? xq : (which == 1) ? xk : xv;
  const bf16_t* WT = (which == 0) ? wtq : (which == 1) ? wtk : wtv;
  const float*  Bb = (which == 0) ? bq : (which == 1) ? bk : bv;

  const int tid = threadIdx.x;
  const int wave = tid >> 5, lane = tid & 31;
  const int lane16 = lane & 15, half = lane >> 4;
  const int m0 = blockIdx.y * 128;
  const int n0 = blockIdx.x * 128;
  const int wm = (wave & 3) * 32;   // wave M offset in tile
  const int wn = (wave >> 2) * 64;  // wave N offset in tile

  fx8 cc[2][4] = {};

#if HAVE_ASYNC
  stage_async(X, WT, m0, n0, 0, tid, As32[0], BsT[0]);
#endif
  int buf = 0;

  for (int kb = 0; kb < EDIM; kb += 32) {
#if HAVE_ASYNC
    if (kb + 32 < EDIM) {
      stage_async(X, WT, m0, n0, kb + 32, tid, As32[buf ^ 1], BsT[buf ^ 1]);
      __builtin_amdgcn_s_wait_asynccnt(6);  // 6 new in flight; old 6 complete
    } else {
      __builtin_amdgcn_s_wait_asynccnt(0);
    }
#else
    stage_sync(X, WT, m0, n0, kb, tid, As32[buf], BsT[buf]);
#endif
    __syncthreads();

    bfx16 af[2];
#pragma unroll
    for (int mi = 0; mi < 2; mi++)
      af[mi] = frag_a_f32(&As32[buf][wm + mi*16 + lane16][half * 8]);
#pragma unroll
    for (int ni = 0; ni < 4; ni++) {
      bfx16 bfr = frag_b(&BsT[buf][wn + ni*16 + lane16][half * 8]);
#pragma unroll
      for (int mi = 0; mi < 2; mi++)
        cc[mi][ni] = wmma_bf16(af[mi], bfr, cc[mi][ni]);
    }
    __syncthreads();
#if HAVE_ASYNC
    buf ^= 1;
#endif
  }

  // Writeout with bias (+ Q scale). C/D layout: M = half*8 + r, N = lane16.
#pragma unroll
  for (int mi = 0; mi < 2; mi++) {
#pragma unroll
    for (int ni = 0; ni < 4; ni++) {
#pragma unroll
      for (int r = 0; r < 8; r++) {
        int m = m0 + wm + mi*16 + half*8 + r;
        int n = n0 + wn + ni*16 + lane16;
        float val = cc[mi][ni][r] + Bb[n];
        int b = m >> 11, s = m & (SEQ - 1);
        int h = n >> 6,  d = n & (DH - 1);
        if (which == 0)
          oq[(size_t)((b*NH + h)*SEQ + s) * DH + d] = (bf16_t)(val * QSCALE);
        else if (which == 1)
          ok[(size_t)((b*NH + h)*SEQ + s) * DH + d] = (bf16_t)val;
        else
          ovt[(size_t)((b*NH + h)*DH + d) * SEQ + s] = (bf16_t)val;
      }
    }
  }
}

// ---------------------------------------------------------------------------
// Kernel 2: logits = (Q/sqrt(Dh)) @ K^T, written f32 into attn region.
// grid = (S/32, B*H), block = 256. Waves partition the key axis.
// ---------------------------------------------------------------------------
__global__ __launch_bounds__(256) void logits_kernel(
    const bf16_t* __restrict__ wsQ, const bf16_t* __restrict__ wsK,
    float* __restrict__ attn)
{
  const int tid = threadIdx.x;
  const int wave = tid >> 5, lane = tid & 31;
  const int lane16 = lane & 15, half = lane >> 4;
  const int q0 = blockIdx.x * 32;
  const int bh = blockIdx.y;

  const bf16_t* Qh = wsQ + (size_t)bh * SEQ * DH;
  const bf16_t* Kh = wsK + (size_t)bh * SEQ * DH;
  float* arow = attn + (size_t)bh * SEQ * SEQ;

  // Q fragments: 2 M-tiles x 2 K-steps (Dh = 64 = 2 x 32)
  bfx16 af[2][2];
#pragma unroll
  for (int mi = 0; mi < 2; mi++)
#pragma unroll
    for (int ks = 0; ks < 2; ks++)
      af[mi][ks] = frag_a(Qh + (size_t)(q0 + mi*16 + lane16) * DH + ks*32 + half*8);

  for (int nt = wave * 16; nt < SEQ; nt += 128) {
    if (nt + 128 < SEQ)
      __builtin_prefetch(Kh + (size_t)(nt + 128 + lane16) * DH, 0, 1);
    fx8 acc[2] = {};
#pragma unroll
    for (int ks = 0; ks < 2; ks++) {
      bfx16 bfr = frag_b(Kh + (size_t)(nt + lane16) * DH + ks*32 + half*8);
      acc[0] = wmma_bf16(af[0][ks], bfr, acc[0]);
      acc[1] = wmma_bf16(af[1][ks], bfr, acc[1]);
    }
#pragma unroll
    for (int mi = 0; mi < 2; mi++)
#pragma unroll
      for (int r = 0; r < 8; r++) {
        int q = q0 + mi*16 + half*8 + r;
        arow[(size_t)q * SEQ + nt + lane16] = acc[mi][r];
      }
  }
}

// ---------------------------------------------------------------------------
// Kernel 3: per-row softmax stats (max, 1/sum(exp)). One wave per row.
// grid = B*H*S/8 blocks, block = 256 (8 waves).
// ---------------------------------------------------------------------------
__global__ __launch_bounds__(256) void softstats_kernel(
    const float* __restrict__ attn, float* __restrict__ stats)
{
  const int tid = threadIdx.x;
  const int wave = tid >> 5, lane = tid & 31;
  const size_t row = (size_t)blockIdx.x * 8 + wave;
  const float* p = attn + row * SEQ;

  float4 vals[16];
#pragma unroll
  for (int c = 0; c < 16; c++)
    vals[c] = *(const float4*)(p + (c * 32 + lane) * 4);

  float m = -3.4e38f;
#pragma unroll
  for (int c = 0; c < 16; c++)
    m = fmaxf(m, fmaxf(fmaxf(vals[c].x, vals[c].y), fmaxf(vals[c].z, vals[c].w)));
#pragma unroll
  for (int off = 16; off > 0; off >>= 1)
    m = fmaxf(m, __shfl_xor(m, off, 32));

  float s = 0.0f;
#pragma unroll
  for (int c = 0; c < 16; c++)
    s += __expf(vals[c].x - m) + __expf(vals[c].y - m)
       + __expf(vals[c].z - m) + __expf(vals[c].w - m);
#pragma unroll
  for (int off = 16; off > 0; off >>= 1)
    s += __shfl_xor(s, off, 32);

  if (lane == 0) {
    stats[row * 2 + 0] = m;
    stats[row * 2 + 1] = 1.0f / s;
  }
}

// ---------------------------------------------------------------------------
// Kernel 4: normalize attn in place + out = attn @ V via WMMA.
// grid = (S/32, B*H), block = 256; 8 waves each own one 16x16 out tile.
// ---------------------------------------------------------------------------
__global__ __launch_bounds__(256) void av_kernel(
    float* __restrict__ attn, const float* __restrict__ stats,
    const bf16_t* __restrict__ wsVt, float* __restrict__ out)
{
  __shared__ alignas(16) bf16_t Ps[32][40];
  __shared__ float sm[32];
  __shared__ float srs[32];

  const int tid = threadIdx.x;
  const int wave = tid >> 5, lane = tid & 31;
  const int lane16 = lane & 15, half = lane >> 4;
  const int q0 = blockIdx.x * 32;
  const int bh = blockIdx.y;
  const int b = bh >> 4, h = bh & 15;

  float* arow = attn + (size_t)bh * SEQ * SEQ;
  const bf16_t* Vth = wsVt + (size_t)bh * DH * SEQ;

  if (tid < 32) {
    size_t row = (size_t)bh * SEQ + q0 + tid;
    sm[tid]  = stats[row * 2 + 0];
    srs[tid] = stats[row * 2 + 1];
  }
  __syncthreads();

  const int mi = wave & 1;   // M sub-tile (0..1)
  const int ni = wave >> 1;  // N sub-tile (0..3)
  fx8 cc = {};

  const int prow = tid >> 3;        // 32 rows, 8 threads/row
  const int pcol = (tid & 7) * 4;   // 4 cols each
  const float pm  = sm[prow];
  const float prs = srs[prow];
  float* pbase = arow + (size_t)(q0 + prow) * SEQ + pcol;

  for (int kc = 0; kc < SEQ; kc += 32) {
    // Cooperative: normalize 32x32 chunk, write attn back, stage bf16 P in LDS
    float4 lv = *(const float4*)(pbase + kc);
    float4 pv;
    pv.x = __expf(lv.x - pm) * prs;
    pv.y = __expf(lv.y - pm) * prs;
    pv.z = __expf(lv.z - pm) * prs;
    pv.w = __expf(lv.w - pm) * prs;
    *(float4*)(pbase + kc) = pv;
    Ps[prow][pcol+0] = (bf16_t)pv.x; Ps[prow][pcol+1] = (bf16_t)pv.y;
    Ps[prow][pcol+2] = (bf16_t)pv.z; Ps[prow][pcol+3] = (bf16_t)pv.w;
    __syncthreads();

    bfx16 af  = frag_a(&Ps[mi*16 + lane16][half * 8]);
    bfx16 bfr = frag_b(Vth + (size_t)(ni*16 + lane16) * SEQ + kc + half*8);
    cc = wmma_bf16(af, bfr, cc);
    __syncthreads();
  }

#pragma unroll
  for (int r = 0; r < 8; r++) {
    int q = q0 + mi*16 + half*8 + r;
    int d = ni*16 + lane16;
    out[(size_t)(b*SEQ + q) * EDIM + h*DH + d] = cc[r];
  }
}

// ---------------------------------------------------------------------------
extern "C" void kernel_launch(void* const* d_in, const int* in_sizes, int n_in,
                              void* d_out, int out_size, void* d_ws, size_t ws_size,
                              hipStream_t stream) {
  (void)in_sizes; (void)n_in; (void)out_size; (void)ws_size;
  const float* q  = (const float*)d_in[0];
  const float* k  = (const float*)d_in[1];
  const float* v  = (const float*)d_in[2];
  const float* wq = (const float*)d_in[3];
  const float* bq = (const float*)d_in[4];
  const float* wk = (const float*)d_in[5];
  const float* bk = (const float*)d_in[6];
  const float* wv = (const float*)d_in[7];
  const float* bv = (const float*)d_in[8];

  float* out  = (float*)d_out;
  float* attn = out + (size_t)BATCH * SEQ * EDIM;   // tuple: (out, attn)

  const size_t HEAD_ELEMS = (size_t)BATCH * NH * SEQ * DH;  // 4,194,304
  const size_t W_ELEMS    = (size_t)EDIM * EDIM;            // 1,048,576
  bf16_t* wsQ   = (bf16_t*)d_ws;
  bf16_t* wsK   = wsQ + HEAD_ELEMS;
  bf16_t* wsVt  = wsK + HEAD_ELEMS;
  float*  stats = (float*)(wsVt + HEAD_ELEMS);      // [B*H*S][2]
  bf16_t* wtq   = (bf16_t*)(stats + (size_t)BATCH * NH * SEQ * 2);
  bf16_t* wtk   = wtq + W_ELEMS;
  bf16_t* wtv   = wtk + W_ELEMS;

  // 0) Weights: f32 [k][n] -> bf16 [n][k]
  convert_w_kernel<<<dim3(EDIM/64, EDIM/64, 3), 256, 0, stream>>>(
      wq, wk, wv, wtq, wtk, wtv);

  // 1) QKV projections (bf16 WMMA, async double-buffered staging)
  qkv_proj_kernel<<<dim3(EDIM/128, (BATCH*SEQ)/128, 3), 256, 0, stream>>>(
      q, k, v, wtq, wtk, wtv, bq, bk, bv, wsQ, wsK, wsVt);

  // 2) Scaled logits into attn region
  logits_kernel<<<dim3(SEQ/32, BATCH*NH), 256, 0, stream>>>(wsQ, wsK, attn);

  // 3) Row softmax stats
  softstats_kernel<<<dim3((BATCH*NH*SEQ)/8), 256, 0, stream>>>(attn, stats);

  // 4) Normalize attn in place + out = attn @ V
  av_kernel<<<dim3(SEQ/32, BATCH*NH), 256, 0, stream>>>(attn, stats, wsVt, out);
}
